// Attention_21088289423660
// MI455X (gfx1250) — matrix operand order repaired and checked by reading, not compiled, not run on hardware
//
#include <hip/hip_runtime.h>
#include <hip/hip_bf16.h>

// ---------------- problem constants ----------------
#define DIMF   256
#define HIDF   128
#define BSZ    4
#define NNT    4096
#define MMT    8
#define NMTOK  (NNT*MMT)        // 32768 tokens per batch
#define TTOK   (BSZ*NMTOK)      // 131072 tokens total
#define NTILES (TTOK/16)        // 8192 row tiles of 16 tokens
#define WSTR   264              // padded LDS stride (K=256) -> 16B aligned, 2-way banks
#define WSTR2  136              // padded LDS stride (K=128)

typedef _Float16 v8h  __attribute__((ext_vector_type(8)));
typedef _Float16 v16h __attribute__((ext_vector_type(16)));
typedef float    v8f  __attribute__((ext_vector_type(8)));
typedef unsigned int u32x4 __attribute__((ext_vector_type(4)));
typedef int          i32x4 __attribute__((ext_vector_type(4)));
typedef int          i32x8 __attribute__((ext_vector_type(8)));

union U16h { v16h v; v8h h[2]; };

// A fragment 16x32 f16: lane<16 -> row m=lane, K = kb+[0..7], kb+[16..23]
//                       lane>=16 -> row m=lane-16, K = kb+[8..15], kb+[24..31]
__device__ __forceinline__ v16h ldAfrag(const _Float16* base, int stride, int kb, int lane) {
  const int m = lane & 15, hf = lane >> 4;
  const _Float16* p = base + (size_t)m * stride + kb + hf * 8;
  U16h u;
  u.h[0] = *(const v8h*)(p);
  u.h[1] = *(const v8h*)(p + 16);
  return u.v;
}

// B fragment 32x16 f16 from W_T[n][k]: lane<16 -> col n=nb+lane, K=kb+[0..7], kb+[16..23]
//                                      lane>=16 -> col n=nb+lane-16, K=kb+[8..15], kb+[24..31]
__device__ __forceinline__ v16h ldBfrag(const _Float16* base, int stride, int nb, int kb, int lane) {
  const int n = lane & 15, hf = lane >> 4;
  const _Float16* p = base + (size_t)(nb * 16 + n) * stride + kb + hf * 8;
  U16h u;
  u.h[0] = *(const v8h*)(p);
  u.h[1] = *(const v8h*)(p + 16);
  return u.v;
}

__device__ __forceinline__ v8f wmma32(v16h a, v16h b, v8f c) {
  return __builtin_amdgcn_wmma_f32_16x16x32_f16(false, a, false, b, (short)0, c, false, false);
}

__device__ __forceinline__ void wait_ds() {
  asm volatile("s_wait_dscnt 0" ::: "memory");
}

// ---- Tensor Data Mover: DMA a rows x kdim f16 matrix (row-major, dense in
// global) into LDS, inserting pad bytes each row via D#.pad so the LDS image
// has the bank-conflict-free stride (WSTR/WSTR2) expected by ldBfrag.
// padCode: pad_interval code (6 -> every 128 DWORDs = 512B row; 5 -> 256B row)
// pad amount fixed at code 3 = 4 DWORDs = 16 bytes.
__device__ __forceinline__ void tdm_load_w(_Float16* dstLds, const _Float16* src,
                                           unsigned rows, unsigned kdim,
                                           unsigned padCode) {
  const unsigned long long ga = (unsigned long long)src;
  const unsigned lds = (unsigned)(unsigned long long)(void*)dstLds; // low 32 = LDS offset
  u32x4 g0;
  g0[0] = 1u;                                        // count=1 (valid), user mode
  g0[1] = lds;                                       // lds_addr
  g0[2] = (unsigned)ga;                              // global_addr[31:0]
  g0[3] = (unsigned)(ga >> 32) | (2u << 30);         // global_addr[56:32] | type=2
  i32x8 g1;
  g1[0] = (int)((1u << 16)                           // data_size = 2 bytes
              | (1u << 20)                           // pad_enable
              | (padCode << 22)                      // pad_interval
              | (3u << 25));                         // pad_amount = 4 DWORDs (16B)
  g1[1] = (int)(kdim << 16);                         // abar_addr=0 | tensor_dim0 lo16
  g1[2] = (int)((kdim >> 16) | (rows << 16));        // tensor_dim0 hi16 | tensor_dim1 lo16
  g1[3] = (int)((rows >> 16) | (kdim << 16));        // tensor_dim1 hi16 | tile_dim0
  g1[4] = (int)rows;                                 // tile_dim1 | tile_dim2=0
  g1[5] = (int)kdim;                                 // tensor_dim0_stride lo32
  g1[6] = 0;                                         // stride0 hi16 | stride1 lo16
  g1[7] = 0;                                         // stride1 hi32
  const i32x4 z4 = {0, 0, 0, 0};
#if defined(__clang_major__) && (__clang_major__ >= 23)
  const i32x8 z8 = {0, 0, 0, 0, 0, 0, 0, 0};
  __builtin_amdgcn_tensor_load_to_lds(g0, g1, z4, z4, z8, 0);
#else
  __builtin_amdgcn_tensor_load_to_lds(g0, g1, z4, z4, 0);
#endif
}

// ---------------- K0: weight transpose + f32->f16 ----------------
__global__ void k_wconv(const float* __restrict__ src, _Float16* __restrict__ dst,
                        int K, int Nn) {
  const int idx = blockIdx.x * blockDim.x + threadIdx.x;
  if (idx >= K * Nn) return;
  const int kk = idx / Nn;
  const int n  = idx - kk * Nn;
  dst[(size_t)n * K + kk] = (_Float16)src[idx];   // W_T[n][k] = W[k][n]
}

// ---------------- K1: kh = k@Wk ; s = kh - q@Wq + pos ----------------
__global__ __launch_bounds__(192) void k_proj(
    const float* __restrict__ q, const float* __restrict__ k,
    const float* __restrict__ pos,
    const _Float16* __restrict__ WqT, const _Float16* __restrict__ WkT,
    _Float16* __restrict__ khB, _Float16* __restrict__ sB)
{
  __shared__ _Float16 sWk[256 * WSTR];       // 135168 B
  __shared__ _Float16 sWq[256 * WSTR];       // 135168 B
  __shared__ _Float16 sA[6][16 * WSTR];      //  50688 B  (total 321024 <= 320KB)

  const int tid = threadIdx.x;
  if (tid < 32) {                            // wave 0 issues TDM weight DMAs
    tdm_load_w(sWk, WkT, 256, 256, 6);
    tdm_load_w(sWq, WqT, 256, 256, 6);
    __builtin_amdgcn_s_wait_tensorcnt(0);
  }
  __syncthreads();

  const int lane = tid & 31;
  const int wv   = tid >> 5;
  const int hf   = lane >> 4;
  const int nn   = lane & 15;

  for (int tile = blockIdx.x * 6 + wv; tile < NTILES; tile += gridDim.x * 6) {
    const int tokbase = tile * 16;
    const int bidx = tokbase / NMTOK;
    const int rem  = tokbase - bidx * NMTOK;            // multiple of 16
    const size_t gb = (size_t)bidx * DIMF * NMTOK + rem;
    _Float16* A = sA[wv];

    v16h kf[8], qf[8];

    // ---- stage k tile (16 tok x 256 d), d-major strided gather ----
    {
      const float* src = k + gb;
      #pragma unroll
      for (int i = 0; i < 32; i++) {
        const int d  = i * 8 + (lane >> 2);
        const int t4 = (lane & 3) * 4;
        const float4 f = *(const float4*)(src + (size_t)d * NMTOK + t4);
        A[(t4 + 0) * WSTR + d] = (_Float16)f.x;
        A[(t4 + 1) * WSTR + d] = (_Float16)f.y;
        A[(t4 + 2) * WSTR + d] = (_Float16)f.z;
        A[(t4 + 3) * WSTR + d] = (_Float16)f.w;
      }
      wait_ds();
      #pragma unroll
      for (int ks = 0; ks < 8; ks++) kf[ks] = ldAfrag(A, WSTR, ks * 32, lane);
    }
    // ---- stage -q tile (negation folds the kh - qh subtraction) ----
    {
      const float* src = q + gb;
      #pragma unroll
      for (int i = 0; i < 32; i++) {
        const int d  = i * 8 + (lane >> 2);
        const int t4 = (lane & 3) * 4;
        const float4 f = *(const float4*)(src + (size_t)d * NMTOK + t4);
        A[(t4 + 0) * WSTR + d] = (_Float16)(-f.x);
        A[(t4 + 1) * WSTR + d] = (_Float16)(-f.y);
        A[(t4 + 2) * WSTR + d] = (_Float16)(-f.z);
        A[(t4 + 3) * WSTR + d] = (_Float16)(-f.w);
      }
      wait_ds();
      #pragma unroll
      for (int ks = 0; ks < 8; ks++) qf[ks] = ldAfrag(A, WSTR, ks * 32, lane);
    }

    // ---- 16 column tiles; acc chain kh -> s ----
    for (int nb = 0; nb < 16; nb++) {
      v8f ak = {0.f,0.f,0.f,0.f,0.f,0.f,0.f,0.f};
      #pragma unroll
      for (int ks = 0; ks < 8; ks++)
        ak = wmma32(kf[ks], ldBfrag(sWk, WSTR, nb, ks * 32, lane), ak);
      v8f as = ak;
      #pragma unroll
      for (int ks = 0; ks < 8; ks++)
        as = wmma32(qf[ks], ldBfrag(sWq, WSTR, nb, ks * 32, lane), as);

      const int c = nb * 16 + nn;
      #pragma unroll
      for (int r = 0; r < 8; r++) {
        const int m   = r + 8 * hf;
        const int tok = tokbase + m;
        const float pv = pos[(size_t)tok * DIMF + c];
        khB[(size_t)tok * DIMF + c] = (_Float16)ak[r];
        sB [(size_t)tok * DIMF + c] = (_Float16)(as[r] + pv);
      }
    }
  }
}

// ---------------- K2: vp = kh@Wv + pos ; h = relu(s@W1 + b1) ----------------
__global__ __launch_bounds__(256) void k_vh(
    const _Float16* __restrict__ khB, const _Float16* __restrict__ sB,
    const float* __restrict__ pos, const float* __restrict__ b1,
    const _Float16* __restrict__ WvT, const _Float16* __restrict__ W1T,
    _Float16* __restrict__ vpB, _Float16* __restrict__ hB)
{
  __shared__ _Float16 sWv[256 * WSTR];   // 135168 B
  __shared__ _Float16 sW1[128 * WSTR];   //  67584 B

  const int tid = threadIdx.x;
  if (tid < 32) {
    tdm_load_w(sWv, WvT, 256, 256, 6);
    tdm_load_w(sW1, W1T, 128, 256, 6);
    __builtin_amdgcn_s_wait_tensorcnt(0);
  }
  __syncthreads();

  const int lane = tid & 31;
  const int wv   = tid >> 5;
  const int hf   = lane >> 4;
  const int nn   = lane & 15;

  for (int tile = blockIdx.x * 8 + wv; tile < NTILES; tile += gridDim.x * 8) {
    const int tokbase = tile * 16;

    v16h af[8];
    // A frags straight from global f16 (row-major, 2 x b128 per frag)
    #pragma unroll
    for (int ks = 0; ks < 8; ks++)
      af[ks] = ldAfrag(khB + (size_t)tokbase * DIMF, DIMF, ks * 32, lane);

    for (int nb = 0; nb < 16; nb++) {          // v = kh@Wv, vp = v + pos
      v8f acc = {0.f,0.f,0.f,0.f,0.f,0.f,0.f,0.f};
      #pragma unroll
      for (int ks = 0; ks < 8; ks++)
        acc = wmma32(af[ks], ldBfrag(sWv, WSTR, nb, ks * 32, lane), acc);
      const int c = nb * 16 + nn;
      #pragma unroll
      for (int r = 0; r < 8; r++) {
        const int tok = tokbase + r + 8 * hf;
        const float pv = pos[(size_t)tok * DIMF + c];
        vpB[(size_t)tok * DIMF + c] = (_Float16)(acc[r] + pv);
      }
    }

    #pragma unroll
    for (int ks = 0; ks < 8; ks++)
      af[ks] = ldAfrag(sB + (size_t)tokbase * DIMF, DIMF, ks * 32, lane);

    for (int nb = 0; nb < 8; nb++) {           // h = relu(s@W1 + b1), 128 wide
      v8f acc = {0.f,0.f,0.f,0.f,0.f,0.f,0.f,0.f};
      #pragma unroll
      for (int ks = 0; ks < 8; ks++)
        acc = wmma32(af[ks], ldBfrag(sW1, WSTR, nb, ks * 32, lane), acc);
      const int c = nb * 16 + nn;
      const float bias = b1[c];
      #pragma unroll
      for (int r = 0; r < 8; r++) {
        const int tok = tokbase + r + 8 * hf;
        float u = acc[r] + bias;
        u = u > 0.f ? u : 0.f;
        hB[(size_t)tok * HIDF + c] = (_Float16)u;
      }
    }
  }
}

// ------- K3: attn = sigmoid(mask(h@W2+b2)); out = ((vp)*attn)@Wo + bo -------
__global__ __launch_bounds__(256) void k_attn_out(
    const _Float16* __restrict__ hB, const _Float16* __restrict__ vpB,
    const int* __restrict__ mask,
    const float* __restrict__ b2, const float* __restrict__ bo,
    const _Float16* __restrict__ W2T, const _Float16* __restrict__ WoT,
    float* __restrict__ out)
{
  __shared__ _Float16 sW2[256 * WSTR2];     //  69632 B
  __shared__ _Float16 sWo[256 * WSTR];      // 135168 B
  __shared__ _Float16 sG[8][16 * WSTR];     //  67584 B  gated activations per wave
  __shared__ float    sO[8][16 * 17];       //   8704 B  output transpose per wave

  const int tid = threadIdx.x;
  if (tid < 32) {
    tdm_load_w(sW2, W2T, 256, 128, 5);      // 256B rows -> interval code 5
    tdm_load_w(sWo, WoT, 256, 256, 6);
    __builtin_amdgcn_s_wait_tensorcnt(0);
  }
  __syncthreads();

  const int lane = tid & 31;
  const int wv   = tid >> 5;
  const int hf   = lane >> 4;
  const int nn   = lane & 15;

  for (int tile = blockIdx.x * 8 + wv; tile < NTILES; tile += gridDim.x * 8) {
    const int tokbase = tile * 16;
    const int bidx = tokbase / NMTOK;
    const int rem  = tokbase - bidx * NMTOK;

    v16h hfr[4];
    #pragma unroll
    for (int ks = 0; ks < 4; ks++)
      hfr[ks] = ldAfrag(hB + (size_t)tokbase * HIDF, HIDF, ks * 32, lane);

    _Float16* G = sG[wv];
    for (int nb = 0; nb < 16; nb++) {        // u = h@W2 + b2 -> sigmoid -> gate
      v8f acc = {0.f,0.f,0.f,0.f,0.f,0.f,0.f,0.f};
      #pragma unroll
      for (int ks = 0; ks < 4; ks++)
        acc = wmma32(hfr[ks], ldBfrag(sW2, WSTR2, nb, ks * 32, lane), acc);
      const int c = nb * 16 + nn;
      const float bias = b2[c];
      #pragma unroll
      for (int r = 0; r < 8; r++) {
        const int m   = r + 8 * hf;
        const int tok = tokbase + m;
        float u = acc[r] + bias;
        u = (mask[tok] == 0) ? -1e9f : u;
        const float a = 1.f / (1.f + __expf(-u));
        const float vpv = (float)vpB[(size_t)tok * DIMF + c];
        G[m * WSTR + c] = (_Float16)(vpv * a);
      }
    }
    wait_ds();

    v16h gf[8];
    #pragma unroll
    for (int ks = 0; ks < 8; ks++) gf[ks] = ldAfrag(G, WSTR, ks * 32, lane);

    float* O = sO[wv];
    const size_t ob = (size_t)bidx * DIMF * NMTOK + rem;
    for (int nb = 0; nb < 16; nb++) {        // final GEMM + transposed store
      v8f acc = {0.f,0.f,0.f,0.f,0.f,0.f,0.f,0.f};
      #pragma unroll
      for (int ks = 0; ks < 8; ks++)
        acc = wmma32(gf[ks], ldBfrag(sWo, WSTR, nb, ks * 32, lane), acc);
      const float bias = bo[nb * 16 + nn];
      #pragma unroll
      for (int r = 0; r < 8; r++)
        O[(r + 8 * hf) * 17 + nn] = acc[r] + bias;   // [token][d] in LDS
      wait_ds();
      // write d-major output: lanes 0..15 cover 16 consecutive tokens (64B runs)
      #pragma unroll
      for (int i = 0; i < 8; i++) {
        const int dl = i * 2 + hf;
        out[ob + (size_t)(nb * 16 + dl) * NMTOK + nn] = O[nn * 17 + dl];
      }
      wait_ds();
    }
  }
}

// ---------------- host side ----------------
extern "C" void kernel_launch(void* const* d_in, const int* in_sizes, int n_in,
                              void* d_out, int out_size, void* d_ws, size_t ws_size,
                              hipStream_t stream) {
  const float* q   = (const float*)d_in[0];
  const float* k   = (const float*)d_in[1];
  const float* pos = (const float*)d_in[2];
  const int*   msk = (const int*)d_in[3];
  const float* Wq  = (const float*)d_in[4];
  const float* Wk  = (const float*)d_in[5];
  const float* Wv  = (const float*)d_in[6];
  const float* W1  = (const float*)d_in[7];
  const float* b1  = (const float*)d_in[8];
  const float* W2  = (const float*)d_in[9];
  const float* b2  = (const float*)d_in[10];
  const float* Wo  = (const float*)d_in[11];
  const float* bo  = (const float*)d_in[12];

  char* ws = (char*)d_ws;
  const size_t OFF_WQT = 0;
  const size_t OFF_WKT = OFF_WQT + 256*256*2;
  const size_t OFF_WVT = OFF_WKT + 256*256*2;
  const size_t OFF_W1T = OFF_WVT + 256*256*2;
  const size_t OFF_W2T = OFF_W1T + 256*128*2;
  const size_t OFF_WOT = OFF_W2T + 128*256*2;
  const size_t OFF_KH  = OFF_WOT + 256*256*2;
  const size_t OFF_S   = OFF_KH + (size_t)TTOK*DIMF*2;
  const size_t OFF_VP  = OFF_S  + (size_t)TTOK*DIMF*2;
  const size_t OFF_H   = OFF_VP + (size_t)TTOK*DIMF*2;

  _Float16* WqT = (_Float16*)(ws + OFF_WQT);
  _Float16* WkT = (_Float16*)(ws + OFF_WKT);
  _Float16* WvT = (_Float16*)(ws + OFF_WVT);
  _Float16* W1T = (_Float16*)(ws + OFF_W1T);
  _Float16* W2T = (_Float16*)(ws + OFF_W2T);
  _Float16* WoT = (_Float16*)(ws + OFF_WOT);
  _Float16* khB = (_Float16*)(ws + OFF_KH);
  _Float16* sB  = (_Float16*)(ws + OFF_S);
  _Float16* vpB = (_Float16*)(ws + OFF_VP);
  _Float16* hB  = (_Float16*)(ws + OFF_H);

  k_wconv<<<256, 256, 0, stream>>>(Wq, WqT, 256, 256);
  k_wconv<<<256, 256, 0, stream>>>(Wk, WkT, 256, 256);
  k_wconv<<<256, 256, 0, stream>>>(Wv, WvT, 256, 256);
  k_wconv<<<128, 256, 0, stream>>>(W1, W1T, 256, 128);
  k_wconv<<<128, 256, 0, stream>>>(W2, W2T, 128, 256);
  k_wconv<<<256, 256, 0, stream>>>(Wo, WoT, 256, 256);

  k_proj    <<<512, 192, 0, stream>>>(q, k, pos, WqT, WkT, khB, sB);
  k_vh      <<<512, 256, 0, stream>>>(khB, sB, pos, b1, WvT, W1T, vpB, hB);
  k_attn_out<<<512, 256, 0, stream>>>(hB, vpB, msk, b2, bo, W2T, WoT, (float*)d_out);
}
